// SSIM_78692390797616
// MI455X (gfx1250) — compile-verified, matrix operand order checked by reading, not run on hardware
//
#include <hip/hip_runtime.h>

typedef __attribute__((ext_vector_type(16))) _Float16 v16h;
typedef __attribute__((ext_vector_type(8)))  float    v8f;

#define IMG 512
#define OUTW 502
#define SSIM_C1 0.0001f
#define SSIM_C2 0.0009f

// One wave = one 16x16 output tile of one (n,c) image plane.
// Separable 11-tap Gaussian conv done as two WMMA matmuls against a banded
// constant matrix W (same register image works as A-operand and B-operand
// under the CDNA5 16-bit WMMA layouts).
__global__ __launch_bounds__(32) void ssim_wmma_kernel(
    const float* __restrict__ X, const float* __restrict__ Y,
    float* __restrict__ acc)
{
  const int lane = threadIdx.x;      // 0..31
  const int hi   = lane >> 4;        // 0 = lanes 0-15, 1 = lanes 16-31
  const int m    = lane & 15;        // row (A) / column (B,C,D) index
  const int tx   = blockIdx.x & 31;
  const int ty   = blockIdx.x >> 5;
  const long imgOff = (long)blockIdx.y * (long)(IMG * IMG);
  const float* __restrict__ xi = X + imgOff;
  const float* __restrict__ yi = Y + imgOff;
  const int col0 = tx * 16;
  const int row0 = ty * 16;
  const int coff = hi ? 8 : 0;

  // ---- banded Gaussian matrix W (f16), W[i,j] = g[j-i] in-band else 0 ----
  // (lane,element)->K mapping identical for 16-bit A and B operand layouts:
  //   e<8 : K = e + 8*hi      ; e>=8 : K = e + 8 + 8*hi
  v16h W;
#pragma unroll
  for (int e = 0; e < 16; ++e) {
    const int K = (e < 8 ? e : e + 8) + coff;
    const int d = K - m;
    float wv = 0.0f;
    if (d >= 0 && d <= 10) {
      const float t = (float)(d - 5);
      // gaussian(sigma=1.5), normalized: 1/sum = 0.26600868
      wv = __expf(t * t * (-1.0f / 4.5f)) * 0.26600868f;
    }
    W[e] = (_Float16)wv;
  }

  // ---- zero-padded guarded loads of x,y; element e maps to input col K ----
  float xv[2][16], yv[2][16];
#pragma unroll
  for (int g = 0; g < 2; ++g) {
    const int r = row0 + g * 16 + m;
    const bool rok = r < IMG;
    const float* xr = xi + (long)r * IMG;
    const float* yr = yi + (long)r * IMG;
#pragma unroll
    for (int e = 0; e < 16; ++e) {
      const int c = col0 + (e < 8 ? e : e + 8) + coff;
      const bool ok = rok && (c < IMG);
      xv[g][e] = ok ? xr[c] : 0.0f;
      yv[g][e] = ok ? yr[c] : 0.0f;
    }
  }

  const v8f zero = {};
  v8f Dm[5];   // conv(x), conv(y), conv(x*x), conv(y*y), conv(x*y)

#pragma unroll
  for (int map = 0; map < 5; ++map) {
    // A operands for the two 16-row groups of the horizontal pass
    v16h A0, A1;
#pragma unroll
    for (int e = 0; e < 16; ++e) {
      float v0, v1;
      if      (map == 0) { v0 = xv[0][e];              v1 = xv[1][e]; }
      else if (map == 1) { v0 = yv[0][e];              v1 = yv[1][e]; }
      else if (map == 2) { v0 = xv[0][e] * xv[0][e];   v1 = xv[1][e] * xv[1][e]; }
      else if (map == 3) { v0 = yv[0][e] * yv[0][e];   v1 = yv[1][e] * yv[1][e]; }
      else               { v0 = xv[0][e] * yv[0][e];   v1 = xv[1][e] * yv[1][e]; }
      A0[e] = (_Float16)v0;
      A1[e] = (_Float16)v1;
    }

    // horizontal pass: tmp = A_in x W   (W acts as banded B matrix)
    v8f t0 = __builtin_amdgcn_wmma_f32_16x16x32_f16(false, A0, false, W,
                                                    (short)0, zero, false, false);
    v8f t1 = __builtin_amdgcn_wmma_f32_16x16x32_f16(false, A1, false, W,
                                                    (short)0, zero, false, false);

    // D layout of t0/t1 lines up lane-for-lane with the B layout needed for
    // the vertical pass: pure per-lane f32->f16 repack, no shuffles.
    v16h Bv;
#pragma unroll
    for (int e = 0; e < 8; ++e) {
      Bv[e]     = (_Float16)t0[e];
      Bv[e + 8] = (_Float16)t1[e];
    }

    // vertical pass: out = W x tmp   (W acts as banded A matrix)
    Dm[map] = __builtin_amdgcn_wmma_f32_16x16x32_f16(false, W, false, Bv,
                                                     (short)0, zero, false, false);
  }

  // ---- elementwise SSIM + masked tile sum ----
  float sum = 0.0f;
  const int c = col0 + m;
#pragma unroll
  for (int i = 0; i < 8; ++i) {
    const int r = row0 + i + (hi ? 8 : 0);
    const float mu1 = Dm[0][i];
    const float mu2 = Dm[1][i];
    const float s11 = Dm[2][i] - mu1 * mu1;
    const float s22 = Dm[3][i] - mu2 * mu2;
    const float s12 = Dm[4][i] - mu1 * mu2;
    const float num = (2.0f * mu1 * mu2 + SSIM_C1) * (2.0f * s12 + SSIM_C2);
    const float den = (mu1 * mu1 + mu2 * mu2 + SSIM_C1) * (s11 + s22 + SSIM_C2);
    const float v = num / den;
    sum += (r < OUTW && c < OUTW) ? v : 0.0f;
  }

  // ---- wave reduction (LDS, fixed order) + global atomic ----
  __shared__ float red[32];
  red[lane] = sum;
  __syncthreads();
  if (lane == 0) {
    float s = 0.0f;
#pragma unroll
    for (int i = 0; i < 32; ++i) s += red[i];
    atomicAdd(acc, s);
  }
}

__global__ __launch_bounds__(32) void ssim_zero_kernel(float* __restrict__ acc) {
  if (threadIdx.x == 0) acc[0] = 0.0f;
}

__global__ __launch_bounds__(32) void ssim_scale_kernel(float* __restrict__ acc) {
  if (threadIdx.x == 0) {
    // mean over 16*3*502*502 = 12096192 valid SSIM pixels
    acc[0] = acc[0] * (1.0f / 12096192.0f);
  }
}

extern "C" void kernel_launch(void* const* d_in, const int* in_sizes, int n_in,
                              void* d_out, int out_size, void* d_ws, size_t ws_size,
                              hipStream_t stream) {
  (void)in_sizes; (void)n_in; (void)out_size; (void)d_ws; (void)ws_size;
  const float* X = (const float*)d_in[0];   // "output"
  const float* Y = (const float*)d_in[1];   // "target"
  float* out = (float*)d_out;               // single float (mean SSIM)

  ssim_zero_kernel<<<1, 32, 0, stream>>>(out);

  // 32x32 tiles of 16x16 outputs per plane, 16*3 = 48 planes
  dim3 grid(32 * 32, 48, 1);
  ssim_wmma_kernel<<<grid, 32, 0, stream>>>(X, Y, out);

  ssim_scale_kernel<<<1, 32, 0, stream>>>(out);
}
